// BiLSTM_CRF_1529008357612
// MI455X (gfx1250) — compile-verified, matrix operand order checked by reading, not run on hardware
//
#include <hip/hip_runtime.h>

// ---------------- types ----------------
typedef __attribute__((ext_vector_type(16))) __bf16 v16bf;
typedef __attribute__((ext_vector_type(8)))  float  v8f;
typedef __attribute__((ext_vector_type(2)))  float  f32x2;

// ---------------- problem constants ----------------
#define T_SEQ 4096
#define D_EMB 256
#define HD    256
#define G4    1024      // 4*HD (gates per direction)
#define NCOL  2048      // gates, both directions
#define HCAT  512       // concat hidden
#define TAG   32
#define START_TAG 30
#define STOP_TAG  31
#define NEG_INF  (-10000.0f)

// ---------------- workspace byte offsets ----------------
#define WS_XBF     ((size_t)0)                       // T*D bf16          = 2 MB
#define WS_WIHBF   ((size_t)2097152)                 // 2048*256 bf16     = 1 MB
#define WS_WHHBF   ((size_t)3145728)                 // 2*1024*256 bf16   = 1 MB
#define WS_WOUTBF  ((size_t)4194304)                 // 32*512 bf16       = 32 KB
#define WS_BSUM    ((size_t)4227072)                 // 2048 f32          = 8 KB
#define WS_XG      ((size_t)4235264)                 // T*2048 f32        = 32 MB
#define WS_HCATBF  ((size_t)37789696)                // T*512 bf16        = 4 MB
#define WS_FEATS   ((size_t)41984000)                // T*32 f32          = 512 KB
#define WS_BPS     ((size_t)42508288)                // T*32 i32          = 512 KB
#define WS_PATHREV ((size_t)43032576)                // T i32             = 16 KB

// ---------------- bf16 helpers (bit manipulation, RNE) ----------------
__device__ __forceinline__ unsigned short f2bf(float f) {
    union { float f; unsigned u; } t; t.f = f;
    unsigned u = t.u;
    return (unsigned short)((u + 0x7FFFu + ((u >> 16) & 1u)) >> 16);
}
__device__ __forceinline__ float bf2f(unsigned short h) {
    union { float f; unsigned u; } t; t.u = ((unsigned)h) << 16;
    return t.f;
}

union FragAB { v16bf v; unsigned u[8]; };
union FragC  { v8f   v; float    f[8]; };

// Load a 16x32 bf16 A/B fragment per the CDNA5 VGPR layout.
// base = row pointer (row-major, K contiguous) already advanced to k0.
// half = lane>>4 : selects K sub-ranges {0..7,16..23} vs {8..15,24..31}.
__device__ __forceinline__ void load_frag(FragAB& fr, const unsigned short* base, int half) {
#pragma unroll
    for (int r = 0; r < 8; ++r) {
        int kk = (r < 4) ? (half * 8 + 2 * r) : (16 + half * 8 + 2 * (r - 4));
        fr.u[r] = *(const unsigned int*)(base + kk);
    }
}

// ---------------- kernel 0: weight conversion + bias sums ----------------
extern "C" __global__ void __launch_bounds__(256)
prep_kernel(const float* __restrict__ Wih_f, const float* __restrict__ Whh_f,
            const float* __restrict__ bih_f, const float* __restrict__ bhh_f,
            const float* __restrict__ Wih_b, const float* __restrict__ Whh_b,
            const float* __restrict__ bih_b, const float* __restrict__ bhh_b,
            const float* __restrict__ W_out,
            unsigned short* __restrict__ wih_bf, unsigned short* __restrict__ whh_bf,
            unsigned short* __restrict__ wout_bf, float* __restrict__ bsum) {
    const int N_WIH = NCOL * D_EMB;          // 524288 (both dirs, gate rows 0..2047)
    const int N_WHH = 2 * G4 * HD;           // 524288
    const int N_WOUT = TAG * HCAT;           // 16384
    const int TOTAL = N_WIH + N_WHH + N_WOUT + NCOL;
    int stride = gridDim.x * blockDim.x;
    for (int i = blockIdx.x * blockDim.x + threadIdx.x; i < TOTAL; i += stride) {
        if (i < N_WIH) {
            int row = i >> 8, col = i & 255;
            float v = (row < G4) ? Wih_f[row * D_EMB + col]
                                 : Wih_b[(row - G4) * D_EMB + col];
            wih_bf[i] = f2bf(v);
        } else if (i < N_WIH + N_WHH) {
            int i2 = i - N_WIH;
            int dir = i2 >> 18;              // 262144 per direction
            int w = i2 & 262143;
            whh_bf[i2] = f2bf(dir ? Whh_b[w] : Whh_f[w]);
        } else if (i < N_WIH + N_WHH + N_WOUT) {
            int i3 = i - N_WIH - N_WHH;
            wout_bf[i3] = f2bf(W_out[i3]);
        } else {
            int i4 = i - N_WIH - N_WHH - N_WOUT;
            int dir = i4 >> 10, g = i4 & 1023;
            bsum[i4] = dir ? (bih_b[g] + bhh_b[g]) : (bih_f[g] + bhh_f[g]);
        }
    }
}

// ---------------- kernel 1: embedding gather -> bf16 ----------------
extern "C" __global__ void __launch_bounds__(256)
embed_kernel(const int* __restrict__ seq, const float* __restrict__ E,
             unsigned short* __restrict__ x_bf) {
    int t = blockIdx.x, d = threadIdx.x;
    x_bf[(size_t)t * D_EMB + d] = f2bf(E[(size_t)seq[t] * D_EMB + d]);
}

// ---------------- kernel 2: input projection GEMM (WMMA bf16) ----------
// xg[t, 0:2048] = x[t,:] @ WihC^T + bsum
// Each wave: one 16-row M tile x four 16-col N tiles (A fragment reused 4x).
extern "C" __global__ void __launch_bounds__(256)
inproj_kernel(const unsigned short* __restrict__ x_bf,
              const unsigned short* __restrict__ wih_bf,
              const float* __restrict__ bsum,
              float* __restrict__ xg) {
    int wave = threadIdx.x >> 5;
    int lane = threadIdx.x & 31;
    int lo = lane & 15, half = lane >> 4;
    int gw = blockIdx.x * 8 + wave;          // 8192 waves total
    int m_base = (gw >> 5) * 16;             // 256 m-tiles
    int n_base0 = (gw & 31) * 64;            // 32 groups of 4 n-tiles
    FragC acc[4];
#pragma unroll
    for (int nt = 0; nt < 4; ++nt)
#pragma unroll
        for (int r = 0; r < 8; ++r) acc[nt].f[r] = 0.0f;
    const unsigned short* arow = x_bf + (size_t)(m_base + lo) * D_EMB;
    const unsigned short* brow = wih_bf + (size_t)(n_base0 + lo) * D_EMB;
#pragma unroll
    for (int k0 = 0; k0 < D_EMB; k0 += 32) {
        FragAB a;
        load_frag(a, arow + k0, half);
#pragma unroll
        for (int nt = 0; nt < 4; ++nt) {
            FragAB b;
            load_frag(b, brow + (size_t)(nt * 16) * D_EMB + k0, half);
            acc[nt].v = __builtin_amdgcn_wmma_f32_16x16x32_bf16(
                false, a.v, false, b.v, (short)0, acc[nt].v, false, false);
        }
    }
#pragma unroll
    for (int nt = 0; nt < 4; ++nt) {
        int n = n_base0 + nt * 16 + lo;
        float bv = bsum[n];
#pragma unroll
        for (int r = 0; r < 8; ++r) {
            int m = m_base + r + 8 * half;
            xg[(size_t)m * NCOL + n] = acc[nt].f[r] + bv;
        }
    }
}

// ---------------- kernel 3: LSTM recurrence (one WG per direction) -----
// h kept as f32 in LDS (no convert on load); bf16 weight pairs unpacked with
// shift/and and accumulated as float2 -> v_pk_fma_f32.
extern "C" __global__ void __launch_bounds__(1024)
lstm_kernel(const float* __restrict__ xg,
            const unsigned short* __restrict__ whh_bf,
            const float* __restrict__ h0, const float* __restrict__ c0,
            unsigned short* __restrict__ hcat_bf) {
    __shared__ __align__(16) float h_sh[HD];
    __shared__ float c_sh[HD];
    __shared__ float g_sh[G4];
    const int dir = blockIdx.x;
    const int j = threadIdx.x;
    if (j < HD) {
        c_sh[j] = c0[dir * HD + j];
        h_sh[j] = h0[dir * HD + j];
    }
    __syncthreads();
    const unsigned int* w32 = (const unsigned int*)(whh_bf + ((size_t)dir * G4 + j) * HD);
    const f32x2* h2 = (const f32x2*)h_sh;
    for (int s = 0; s < T_SEQ; ++s) {
        const int tt = dir ? (T_SEQ - 1 - s) : s;
        if (s + 1 < T_SEQ) {
            int tn = dir ? (T_SEQ - 2 - s) : (s + 1);
            __builtin_prefetch(&xg[(size_t)tn * NCOL + dir * G4 + j], 0, 1);
        }
        f32x2 acc2;
        acc2.x = xg[(size_t)tt * NCOL + dir * G4 + j];
        acc2.y = 0.0f;
#pragma unroll 8
        for (int k = 0; k < HD / 2; ++k) {
            unsigned wv = w32[k];
            union { float f; unsigned u; } wl, wh;
            wl.u = wv << 16;
            wh.u = wv & 0xffff0000u;
            f32x2 w2;
            w2.x = wl.f;
            w2.y = wh.f;
            acc2 += h2[k] * w2;              // v_pk_fma_f32 candidate
        }
        g_sh[j] = acc2.x + acc2.y;
        __syncthreads();
        if (j < HD) {
            float gi = 1.0f / (1.0f + expf(-g_sh[j]));
            float gf = 1.0f / (1.0f + expf(-g_sh[HD + j]));
            float gg = tanhf(g_sh[2 * HD + j]);
            float go = 1.0f / (1.0f + expf(-g_sh[3 * HD + j]));
            float c = gf * c_sh[j] + gi * gg;
            float h = go * tanhf(c);
            c_sh[j] = c;
            h_sh[j] = h;
            hcat_bf[(size_t)tt * HCAT + dir * HD + j] = f2bf(h);
        }
        __syncthreads();
    }
}

// ---------------- kernel 4: output projection (WMMA bf16) --------------
// feats[t, 0:32] = hcat[t,:] @ W_out^T + b_out ; one wave per m-tile, both n-tiles.
extern "C" __global__ void __launch_bounds__(256)
outproj_kernel(const unsigned short* __restrict__ hcat_bf,
               const unsigned short* __restrict__ wout_bf,
               const float* __restrict__ b_out,
               float* __restrict__ feats) {
    int wave = threadIdx.x >> 5;
    int lane = threadIdx.x & 31;
    int lo = lane & 15, half = lane >> 4;
    int m_base = (blockIdx.x * 8 + wave) * 16;   // 256 m-tiles
    FragC acc[2];
#pragma unroll
    for (int nt = 0; nt < 2; ++nt)
#pragma unroll
        for (int r = 0; r < 8; ++r) acc[nt].f[r] = 0.0f;
    const unsigned short* arow = hcat_bf + (size_t)(m_base + lo) * HCAT;
    const unsigned short* brow = wout_bf + (size_t)lo * HCAT;
#pragma unroll
    for (int k0 = 0; k0 < HCAT; k0 += 32) {
        FragAB a;
        load_frag(a, arow + k0, half);
#pragma unroll
        for (int nt = 0; nt < 2; ++nt) {
            FragAB b;
            load_frag(b, brow + (size_t)(nt * 16) * HCAT + k0, half);
            acc[nt].v = __builtin_amdgcn_wmma_f32_16x16x32_bf16(
                false, a.v, false, b.v, (short)0, acc[nt].v, false, false);
        }
    }
#pragma unroll
    for (int nt = 0; nt < 2; ++nt) {
        int n = nt * 16 + lo;
        float bv = b_out[n];
#pragma unroll
        for (int r = 0; r < 8; ++r) {
            int m = m_base + r + 8 * half;
            feats[(size_t)m * TAG + n] = acc[nt].f[r] + bv;
        }
    }
}

// ---------------- kernel 5: Viterbi decode (one wave) ------------------
extern "C" __global__ void __launch_bounds__(32)
viterbi_kernel(const float* __restrict__ feats,
               const float* __restrict__ trans,
               int* __restrict__ bps, int* __restrict__ path_rev,
               float* __restrict__ out) {
    __shared__ float tr[TAG * TAG];
    __shared__ float fv[TAG];
    __shared__ float term[TAG];
    const int n = threadIdx.x;
    for (int i = n; i < TAG * TAG; i += TAG) tr[i] = trans[i];
    fv[n] = (n == START_TAG) ? 0.0f : NEG_INF;
    __syncthreads();
    for (int t = 0; t < T_SEQ; ++t) {
        float best = -3.4e38f;
        int bp = 0;
#pragma unroll
        for (int p = 0; p < TAG; ++p) {
            float sc = fv[p] + tr[n * TAG + p];
            if (sc > best) { best = sc; bp = p; }   // strict > => first argmax
        }
        float nv = best + feats[t * TAG + n];
        bps[t * TAG + n] = bp;
        __syncthreads();
        fv[n] = nv;
        __syncthreads();
    }
    term[n] = fv[n] + tr[STOP_TAG * TAG + n];
    __syncthreads();
    if (n == 0) {
        int best = 0; float bs = term[0];
        for (int p = 1; p < TAG; ++p) if (term[p] > bs) { bs = term[p]; best = p; }
        out[0] = bs;
        int b = best;
        for (int s = 0; s < T_SEQ; ++s) {
            b = bps[(T_SEQ - 1 - s) * TAG + b];
            path_rev[s] = b;
        }
        // path = concat(path_rev[::-1][1:], [best])
        for (int t = 0; t < T_SEQ - 1; ++t)
            out[1 + t] = (float)path_rev[T_SEQ - 2 - t];
        out[T_SEQ] = (float)best;
    }
}

// ---------------- launch ----------------
extern "C" void kernel_launch(void* const* d_in, const int* in_sizes, int n_in,
                              void* d_out, int out_size, void* d_ws, size_t ws_size,
                              hipStream_t stream) {
    const int*   seq    = (const int*)  d_in[0];
    const float* E      = (const float*)d_in[1];
    const float* Wih_f  = (const float*)d_in[2];
    const float* Whh_f  = (const float*)d_in[3];
    const float* bih_f  = (const float*)d_in[4];
    const float* bhh_f  = (const float*)d_in[5];
    const float* Wih_b  = (const float*)d_in[6];
    const float* Whh_b  = (const float*)d_in[7];
    const float* bih_b  = (const float*)d_in[8];
    const float* bhh_b  = (const float*)d_in[9];
    const float* h0     = (const float*)d_in[10];
    const float* c0     = (const float*)d_in[11];
    const float* W_out  = (const float*)d_in[12];
    const float* b_out  = (const float*)d_in[13];
    const float* transi = (const float*)d_in[14];
    float* out = (float*)d_out;

    char* ws = (char*)d_ws;
    unsigned short* x_bf    = (unsigned short*)(ws + WS_XBF);
    unsigned short* wih_bf  = (unsigned short*)(ws + WS_WIHBF);
    unsigned short* whh_bf  = (unsigned short*)(ws + WS_WHHBF);
    unsigned short* wout_bf = (unsigned short*)(ws + WS_WOUTBF);
    float*          bsum    = (float*)         (ws + WS_BSUM);
    float*          xg      = (float*)         (ws + WS_XG);
    unsigned short* hcat_bf = (unsigned short*)(ws + WS_HCATBF);
    float*          feats   = (float*)         (ws + WS_FEATS);
    int*            bps     = (int*)           (ws + WS_BPS);
    int*            pathrev = (int*)           (ws + WS_PATHREV);

    prep_kernel<<<2048, 256, 0, stream>>>(Wih_f, Whh_f, bih_f, bhh_f,
                                          Wih_b, Whh_b, bih_b, bhh_b,
                                          W_out, wih_bf, whh_bf, wout_bf, bsum);
    embed_kernel<<<T_SEQ, 256, 0, stream>>>(seq, E, x_bf);
    inproj_kernel<<<1024, 256, 0, stream>>>(x_bf, wih_bf, bsum, xg);
    lstm_kernel<<<2, 1024, 0, stream>>>(xg, whh_bf, h0, c0, hcat_bf);
    outproj_kernel<<<32, 256, 0, stream>>>(hcat_bf, wout_bf, b_out, feats);
    viterbi_kernel<<<1, 32, 0, stream>>>(feats, transi, bps, pathrev, out);
}
